// GNNPolicy_37108517438117
// MI455X (gfx1250) — compile-verified, matrix operand order checked by reading, not run on hardware
//
#include <hip/hip_runtime.h>
#include <hip/hip_bf16.h>

// ---------------- problem constants (from reference) ----------------
constexpr int NN   = 50000;   // nodes (multiple of 16)
constexpr int EE   = 800000;  // edges
constexpr int BB   = 64;      // graphs
constexpr int FEAT = 9;
constexpr int HH   = 128;
constexpr int AA   = 160;
constexpr float NEG_INF = -1000000000.0f;

typedef __attribute__((ext_vector_type(16))) _Float16 v16h;
typedef __attribute__((ext_vector_type(8)))  float    v8f;

union AFrag { v16h v; unsigned int u[8]; _Float16 h[16]; };
union CFrag { v8f  v; float f[8]; };

// ---------------- small utility kernels ----------------
__global__ void zero_f32(float* __restrict__ p, int n) {
    int i = blockIdx.x * blockDim.x + threadIdx.x;
    if (i < n) p[i] = 0.0f;
}

__global__ void deg_count(const int* __restrict__ dst, float* __restrict__ deg, int ne) {
    int i = blockIdx.x * blockDim.x + threadIdx.x;
    if (i < ne) atomicAdd(&deg[dst[i]], 1.0f);
}

__global__ void make_dis(float* __restrict__ deg, int n) {
    int i = blockIdx.x * blockDim.x + threadIdx.x;
    if (i < n) deg[i] = rsqrtf(1.0f + deg[i]);   // dis = (1+indeg)^-0.5, in place
}

// fp32 weight [K,128] -> f16 B fragments, fragment-ordered:
// Wfrag[((k0/32)*256 + tid)*16 + e]  so each GEMM lane does one 32B vector load.
__global__ void convert_w_frag(const float* __restrict__ W, _Float16* __restrict__ Wfrag,
                               int K, int ksteps) {
    int t = blockIdx.x * blockDim.x + threadIdx.x;   // over ksteps*256
    if (t >= ksteps * 256) return;
    int tid   = t & 255;
    int lane  = tid & 31;
    int wave  = tid >> 5;
    int n     = wave * 16 + (lane & 15);
    int khalf = lane >> 4;
    int k0    = (t >> 8) * 32;
    _Float16* dstp = Wfrag + (size_t)t * 16;
    #pragma unroll
    for (int j = 0; j < 8; ++j) {
        int kloc = ((j < 4) ? (2 * j) : (16 + 2 * (j - 4))) + khalf * 8;
        int ka = k0 + kloc;
        dstp[2 * j]     = (ka     < K) ? (_Float16)W[ka * HH + n]       : (_Float16)0.0f;
        dstp[2 * j + 1] = (ka + 1 < K) ? (_Float16)W[(ka + 1) * HH + n] : (_Float16)0.0f;
    }
}

// ---------------- WMMA GEMM: t = A @ W ; agg = t*dis^2 + bias ----------------
// block: 256 thr (8 waves). block -> 16 output rows; wave w -> cols [16w,16w+16)
template<int KPAD, bool DENSE>
__global__ __launch_bounds__(256)
void gemm_wmma(const float* __restrict__ A,          // [N, DENSE?128:FEAT]
               const _Float16* __restrict__ Wfrag,   // fragment-ordered f16 weights
               const float* __restrict__ bias,       // [128]
               const float* __restrict__ dis,        // [N]
               float* __restrict__ t_out,            // [N,128]
               float* __restrict__ agg_out)          // [N,128]
{
    constexpr int LDSW = 34;                         // 17-bank stride: conflict-free
    __shared__ _Float16 ldsA[16 * LDSW];

    const int tid   = threadIdx.x;
    const int lane  = tid & 31;
    const int wave  = tid >> 5;
    const int row0  = blockIdx.x * 16;
    const int m     = lane & 15;
    const int khalf = lane >> 4;
    const int n     = wave * 16 + m;

    CFrag c;
    #pragma unroll
    for (int r = 0; r < 8; ++r) c.f[r] = 0.0f;

    #pragma unroll
    for (int k0 = 0; k0 < KPAD; k0 += 32) {
        __syncthreads();
        if (DENSE) {
            // each thread: one float2 -> packed half2 store (512 elems = 16x32 tile)
            int e  = tid * 2;
            int r  = e >> 5, cc = e & 31;
            const float2 v = *(const float2*)(A + (long)(row0 + r) * HH + k0 + cc);
            union { _Float16 h[2]; unsigned int u; } pk;
            pk.h[0] = (_Float16)v.x;
            pk.h[1] = (_Float16)v.y;
            *(unsigned int*)(&ldsA[r * LDSW + cc]) = pk.u;
        } else {
            #pragma unroll
            for (int i = 0; i < 2; ++i) {
                int e  = tid + i * 256;
                int r  = e >> 5, cc = e & 31;
                float val = (cc < FEAT) ? A[(long)(row0 + r) * FEAT + cc] : 0.0f;
                ldsA[r * LDSW + cc] = (_Float16)val;
            }
        }
        __syncthreads();

        AFrag a;
        #pragma unroll
        for (int j = 0; j < 8; ++j) {
            // documented 16-bit 16x32 layout: VGPR j holds K pair (kloc even -> 4B aligned)
            int kloc = ((j < 4) ? (2 * j) : (16 + 2 * (j - 4))) + khalf * 8;
            a.u[j] = *(const unsigned int*)(&ldsA[m * LDSW + kloc]);
        }
        const v16h b = *(const v16h*)(Wfrag + (size_t)((k0 >> 5) * 256 + tid) * 16);
        c.v = __builtin_amdgcn_wmma_f32_16x16x32_f16(
                  false, a.v, false, b, (short)0, c.v, false, false);
    }

    // epilogue: fuse self-loop term + bias into agg init (N % 16 == 0: no row guard)
    const float bi = bias[n];
    #pragma unroll
    for (int r = 0; r < 8; ++r) {
        int gr = row0 + r + khalf * 8;               // C layout: VGPR r -> M=r / r+8
        float v = c.f[r];
        float d = dis[gr];
        t_out[(long)gr * HH + n]   = v;
        agg_out[(long)gr * HH + n] = v * d * d + bi;
    }
}

// ---------------- edge scatter: agg[dst] += t[src] * dis[src]*dis[dst] --------
// one wave per edge, lane handles 4 contiguous floats
__global__ __launch_bounds__(256)
void scatter_edges(const int* __restrict__ src, const int* __restrict__ dst,
                   const float* __restrict__ dis, const float* __restrict__ t,
                   float* __restrict__ agg, int nedges)
{
    long gt  = (long)blockIdx.x * 256 + threadIdx.x;
    int  e   = (int)(gt >> 5);
    if (e >= nedges) return;
    int lane = (int)(gt & 31);
    int s = src[e], d = dst[e];
    float w = dis[s] * dis[d];
    const float4 v = *(const float4*)(t + (long)s * HH + lane * 4);
    float* ap = agg + (long)d * HH + lane * 4;
    atomicAdd(ap + 0, v.x * w);
    atomicAdd(ap + 1, v.y * w);
    atomicAdd(ap + 2, v.z * w);
    atomicAdd(ap + 3, v.w * w);
}

// ---------------- layernorm(H=128) + relu, one wave per node -----------------
__global__ __launch_bounds__(256)
void ln_relu(const float* __restrict__ agg, const float* __restrict__ g,
             const float* __restrict__ be, float* __restrict__ out, int nrows)
{
    int node = blockIdx.x * 8 + (threadIdx.x >> 5);
    if (node >= nrows) return;
    int lane = threadIdx.x & 31;
    const float4 v = *(const float4*)(agg + (long)node * HH + lane * 4);
    float s  = v.x + v.y + v.z + v.w;
    float sq = v.x * v.x + v.y * v.y + v.z * v.z + v.w * v.w;
    #pragma unroll
    for (int off = 16; off > 0; off >>= 1) {
        s  += __shfl_xor(s,  off, 32);
        sq += __shfl_xor(sq, off, 32);
    }
    float mu  = s * (1.0f / HH);
    float var = sq * (1.0f / HH) - mu * mu;
    float inv = rsqrtf(var + 1e-5f);
    int i = lane * 4;
    float4 o;
    o.x = fmaxf(0.0f, (v.x - mu) * inv * g[i + 0] + be[i + 0]);
    o.y = fmaxf(0.0f, (v.y - mu) * inv * g[i + 1] + be[i + 1]);
    o.z = fmaxf(0.0f, (v.z - mu) * inv * g[i + 2] + be[i + 2]);
    o.w = fmaxf(0.0f, (v.w - mu) * inv * g[i + 3] + be[i + 3]);
    *(float4*)(out + (long)node * HH + i) = o;
}

// ---------------- pooling: segment mean + max (relu outputs >= 0) ------------
__global__ __launch_bounds__(256)
void pool(const float* __restrict__ h, const int* __restrict__ batch,
          float* __restrict__ msum, int* __restrict__ mmax,
          float* __restrict__ cnt, int nrows)
{
    int node = blockIdx.x * 8 + (threadIdx.x >> 5);
    if (node >= nrows) return;
    int lane = threadIdx.x & 31;
    int b = batch[node];
    const float4 v = *(const float4*)(h + (long)node * HH + lane * 4);
    float* mp = msum + b * HH + lane * 4;
    atomicAdd(mp + 0, v.x); atomicAdd(mp + 1, v.y);
    atomicAdd(mp + 2, v.z); atomicAdd(mp + 3, v.w);
    int* xp = mmax + b * HH + lane * 4;   // non-negative floats: int order == float order
    atomicMax(xp + 0, __float_as_int(v.x));
    atomicMax(xp + 1, __float_as_int(v.y));
    atomicMax(xp + 2, __float_as_int(v.z));
    atomicMax(xp + 3, __float_as_int(v.w));
    if (lane == 0) atomicAdd(&cnt[b], 1.0f);
}

__global__ void emb_fin(const float* __restrict__ msum, const int* __restrict__ mmax,
                        const float* __restrict__ cnt, float* __restrict__ emb)
{
    int t = blockIdx.x * blockDim.x + threadIdx.x;   // over B*H
    if (t >= BB * HH) return;
    int b = t >> 7, i = t & 127;
    float c = fmaxf(cnt[b], 1.0f);
    emb[b * 256 + i]       = msum[t] / c;
    emb[b * 256 + 128 + i] = __int_as_float(mmax[t]);
}

// ---------------- heads ----------------
__global__ __launch_bounds__(128)
void head_hidden(const float* __restrict__ emb,
                 const float* __restrict__ AW1, const float* __restrict__ Ab1,
                 const float* __restrict__ CW1, const float* __restrict__ Cb1,
                 float* __restrict__ hidA, float* __restrict__ hidC)
{
    __shared__ float e[256];
    int b = blockIdx.x, j = threadIdx.x;
    e[j]       = emb[b * 256 + j];
    e[j + 128] = emb[b * 256 + 128 + j];
    __syncthreads();
    float accA = Ab1[j], accC = Cb1[j];
    #pragma unroll 4
    for (int k = 0; k < 256; ++k) {
        accA += e[k] * AW1[k * HH + j];
        accC += e[k] * CW1[k * HH + j];
    }
    hidA[b * HH + j] = tanhf(accA);
    hidC[b * HH + j] = tanhf(accC);
}

__global__ __launch_bounds__(160)
void head_logits(const float* __restrict__ hidA, const float* __restrict__ AW2,
                 const float* __restrict__ Ab2, const unsigned char* __restrict__ mask,
                 float* __restrict__ out)
{
    int b = blockIdx.x, a = threadIdx.x;
    float acc = Ab2[a];
    #pragma unroll 4
    for (int j = 0; j < HH; ++j) acc += hidA[b * HH + j] * AW2[j * AA + a];
    out[b * AA + a] = mask[b * AA + a] ? acc : NEG_INF;
}

__global__ __launch_bounds__(256)
void head_value(const float* __restrict__ hidC, const float* __restrict__ CW2,
                const float* __restrict__ Cb2, float* __restrict__ out)
{
    int b = blockIdx.x * 8 + (threadIdx.x >> 5);
    if (b >= BB) return;
    int lane = threadIdx.x & 31;
    const float4 v = *(const float4*)(hidC + b * HH + lane * 4);
    const float4 w = *(const float4*)(CW2 + lane * 4);
    float s = v.x * w.x + v.y * w.y + v.z * w.z + v.w * w.w;
    #pragma unroll
    for (int off = 16; off > 0; off >>= 1) s += __shfl_xor(s, off, 32);
    if (lane == 0) out[BB * AA + b] = s + Cb2[0];
}

// ---------------- launcher ----------------
extern "C" void kernel_launch(void* const* d_in, const int* in_sizes, int n_in,
                              void* d_out, int out_size, void* d_ws, size_t ws_size,
                              hipStream_t stream)
{
    const float* x     = (const float*)d_in[0];
    const int*   ei    = (const int*)d_in[1];
    const int*   src   = ei;
    const int*   dst   = ei + EE;
    const int*   batch = (const int*)d_in[2];
    const unsigned char* amask = (const unsigned char*)d_in[3];
    const float* W1  = (const float*)d_in[4],  *b1  = (const float*)d_in[5];
    const float* g1  = (const float*)d_in[6],  *be1 = (const float*)d_in[7];
    const float* W2  = (const float*)d_in[8],  *b2  = (const float*)d_in[9];
    const float* g2  = (const float*)d_in[10], *be2 = (const float*)d_in[11];
    const float* W3  = (const float*)d_in[12], *b3  = (const float*)d_in[13];
    const float* g3  = (const float*)d_in[14], *be3 = (const float*)d_in[15];
    const float* AW1 = (const float*)d_in[16], *Ab1 = (const float*)d_in[17];
    const float* AW2 = (const float*)d_in[18], *Ab2 = (const float*)d_in[19];
    const float* CW1 = (const float*)d_in[20], *Cb1 = (const float*)d_in[21];
    const float* CW2 = (const float*)d_in[22], *Cb2 = (const float*)d_in[23];
    float* out = (float*)d_out;

    // workspace carve-up (256B aligned)
    char* wp = (char*)d_ws;
    auto take = [&](size_t bytes) -> char* {
        char* p = wp; wp += (bytes + 255) & ~(size_t)255; return p;
    };
    float*     dis   = (float*)take((size_t)NN * 4);
    float*     tbuf  = (float*)take((size_t)NN * HH * 4);
    float*     agg   = (float*)take((size_t)NN * HH * 4);
    float*     hbuf  = (float*)take((size_t)NN * HH * 4);
    _Float16*  Wfrag = (_Float16*)take((size_t)4 * 256 * 16 * 2);   // up to 4 k-steps
    float*     msum  = (float*)take((size_t)(BB * HH * 2 + BB) * 4); // msum|mmax|cnt
    int*       mmax  = (int*)(msum + BB * HH);
    float*     cnt   = (float*)(msum + 2 * BB * HH);
    float*     emb   = (float*)take((size_t)BB * 256 * 4);
    float*     hidA  = (float*)take((size_t)BB * HH * 4);
    float*     hidC  = (float*)take((size_t)BB * HH * 4);

    // degree -> dis
    zero_f32<<<(NN + 255) / 256, 256, 0, stream>>>(dis, NN);
    deg_count<<<(EE + 255) / 256, 256, 0, stream>>>(dst, dis, EE);
    make_dis<<<(NN + 255) / 256, 256, 0, stream>>>(dis, NN);

    const int gemmGrid    = NN / 16;                 // 3125
    const int scatterGrid = (EE * 32) / 256;         // 100000
    const int nodeGrid    = (NN + 7) / 8;            // 6250

    // ---- layer 1 (K=9 -> pad 32) ----
    convert_w_frag<<<1, 256, 0, stream>>>(W1, Wfrag, FEAT, 1);
    gemm_wmma<32, false><<<gemmGrid, 256, 0, stream>>>(x, Wfrag, b1, dis, tbuf, agg);
    scatter_edges<<<scatterGrid, 256, 0, stream>>>(src, dst, dis, tbuf, agg, EE);
    ln_relu<<<nodeGrid, 256, 0, stream>>>(agg, g1, be1, hbuf, NN);

    // ---- layer 2 ----
    convert_w_frag<<<4, 256, 0, stream>>>(W2, Wfrag, HH, 4);
    gemm_wmma<128, true><<<gemmGrid, 256, 0, stream>>>(hbuf, Wfrag, b2, dis, tbuf, agg);
    scatter_edges<<<scatterGrid, 256, 0, stream>>>(src, dst, dis, tbuf, agg, EE);
    ln_relu<<<nodeGrid, 256, 0, stream>>>(agg, g2, be2, hbuf, NN);

    // ---- layer 3 ----
    convert_w_frag<<<4, 256, 0, stream>>>(W3, Wfrag, HH, 4);
    gemm_wmma<128, true><<<gemmGrid, 256, 0, stream>>>(hbuf, Wfrag, b3, dis, tbuf, agg);
    scatter_edges<<<scatterGrid, 256, 0, stream>>>(src, dst, dis, tbuf, agg, EE);
    ln_relu<<<nodeGrid, 256, 0, stream>>>(agg, g3, be3, hbuf, NN);

    // ---- pooling ----
    zero_f32<<<(BB * HH * 2 + BB + 255) / 256, 256, 0, stream>>>(msum, BB * HH * 2 + BB);
    pool<<<nodeGrid, 256, 0, stream>>>(hbuf, batch, msum, mmax, cnt, NN);
    emb_fin<<<(BB * HH + 255) / 256, 256, 0, stream>>>(msum, mmax, cnt, emb);

    // ---- heads ----
    head_hidden<<<BB, 128, 0, stream>>>(emb, AW1, Ab1, CW1, Cb1, hidA, hidC);
    head_logits<<<BB, AA, 0, stream>>>(hidA, AW2, Ab2, amask, out);
    head_value<<<BB / 8, 256, 0, stream>>>(hidC, CW2, Cb2, out);
}
